// DCausalSelfAttention_47614007443570
// MI455X (gfx1250) — compile-verified, matrix operand order checked by reading, not run on hardware
//
#include <hip/hip_runtime.h>
#include <hip/hip_bf16.h>

// ---------------------------------------------------------------------------
// CDNA5 (gfx1250) causal self-attention, bf16 WMMA with fp32 accumulate.
// B=2, T=2048, C=2048, H=16, D=128.
// TDM (tensor_load_to_lds) stages A-tiles / K-tiles; WMMA does all matmul.
// ---------------------------------------------------------------------------

typedef __bf16 v16bf __attribute__((ext_vector_type(16)));
typedef float  v8f   __attribute__((ext_vector_type(8)));
typedef unsigned int u32x4 __attribute__((ext_vector_type(4)));
typedef int          i32x4 __attribute__((ext_vector_type(4)));
typedef int          i32x8 __attribute__((ext_vector_type(8)));

#define WMMA_BF16(a, b, c) \
  __builtin_amdgcn_wmma_f32_16x16x32_bf16(false, (a), false, (b), (short)0, (c), false, false)

#if defined(__HIP_DEVICE_COMPILE__)
#if __has_builtin(__builtin_amdgcn_tensor_load_to_lds) && \
    __has_builtin(__builtin_amdgcn_s_wait_tensorcnt)
#define HAVE_TDM 1
#endif
#endif

// ---------------------------------------------------------------------------
// TDM: async 2D bf16 tile load global -> LDS (row-major, packed in LDS).
// D# packing per CDNA5 ISA 8.3/8.4: group0 = {flags, lds_addr, gaddr57, type=2},
// group1 = {data_size=2B, tensor_dim0/1, tile_dim0/1, dim0_stride}. 2D tile ->
// groups 2/3 zero. Issued by one wave; completion via s_wait_tensorcnt.
// ---------------------------------------------------------------------------
__device__ __forceinline__ void tdm_load_tile_bf16(unsigned ldsOff, const __bf16* gptr,
                                                   unsigned tileW, unsigned tileH,
                                                   unsigned rowStrideElems,
                                                   unsigned tensorW, unsigned tensorH) {
#if defined(HAVE_TDM)
  unsigned long long ga = (unsigned long long)gptr;
  u32x4 g0 = {1u,                                   // count=1, is_restore=0, gather=0
              ldsOff,                               // lds_addr (bytes)
              (unsigned)ga,                         // global_addr[31:0]
              ((unsigned)((ga >> 32) & 0x01FFFFFFu)) | 0x80000000u};  // addr[56:32] | type=2
  i32x8 g1 = {(int)(1u << 16),                      // workgroup_mask=0, data_size=1 (2B)
              (int)(tensorW << 16),                 // tensor_dim0[15:0] @ bits63:48
              (int)((tensorW >> 16) | (tensorH << 16)),   // dim0 hi | dim1 lo
              (int)((tensorH >> 16) | (tileW << 16)),     // dim1 hi | tile_dim0
              (int)(tileH & 0xFFFFu),               // tile_dim1 (tile_dim2=0)
              (int)rowStrideElems,                  // tensor_dim0_stride[31:0]
              0,                                    // stride0 hi | stride1 lo
              0};
  i32x4 gz = {0, 0, 0, 0};
#if __clang_major__ >= 23
  i32x8 gz8 = {0, 0, 0, 0, 0, 0, 0, 0};
  __builtin_amdgcn_tensor_load_to_lds(g0, g1, gz, gz, gz8, 0);
#else
  __builtin_amdgcn_tensor_load_to_lds(g0, g1, gz, gz, 0);
#endif
#else
  (void)ldsOff; (void)gptr; (void)tileW; (void)tileH;
  (void)rowStrideElems; (void)tensorW; (void)tensorH;
#endif
}

__device__ __forceinline__ void tdm_wait() {
#if defined(HAVE_TDM)
  __builtin_amdgcn_s_wait_tensorcnt(0);
#endif
}

__device__ __forceinline__ unsigned lds_offset_of(const void* p) {
  return (unsigned)(unsigned long long)p;  // low 32 bits of generic ptr = LDS offset
}

union FragU { v16bf v; uint4 q[2]; };

// 16x32 bf16 WMMA fragment from a row-major buffer: lane&15 = row, lane>>4
// = h picks k 8h..8h+7 and 16+8h..16+8h+7; K elements are two contiguous 16-byte chunks.
__device__ __forceinline__ v16bf load_frag(const __bf16* base, int ld) {
  const int lane = threadIdx.x & 31;
  const int row  = lane & 15;
  const int half = (lane >> 4) & 1;
  const __bf16* p = base + row * ld + half * 8;
  FragU f;
  f.q[0] = *(const uint4*)(p);
  f.q[1] = *(const uint4*)(p + 16);
  return f.v;
}

__device__ __forceinline__ float hmax16(float v) {
#pragma unroll
  for (int m = 8; m >= 1; m >>= 1) v = fmaxf(v, __shfl_xor(v, m, 32));
  return v;
}
__device__ __forceinline__ float hsum16(float v) {
#pragma unroll
  for (int m = 8; m >= 1; m >>= 1) v += __shfl_xor(v, m, 32);
  return v;
}

// ---------------------------------------------------------------------------
// fp32 -> bf16 conversion, 8 elements per thread (16B store)
// ---------------------------------------------------------------------------
__global__ void __launch_bounds__(256) k_f32_to_bf16(const float* __restrict__ src,
                                                     __bf16* __restrict__ dst, long n) {
  long i = ((long)blockIdx.x * blockDim.x + threadIdx.x) * 8;
  if (i >= n) return;
  float4 a = *(const float4*)(src + i);
  float4 b = *(const float4*)(src + i + 4);
  union { __bf16 h[8]; uint4 u; } o;
  o.h[0] = (__bf16)a.x; o.h[1] = (__bf16)a.y; o.h[2] = (__bf16)a.z; o.h[3] = (__bf16)a.w;
  o.h[4] = (__bf16)b.x; o.h[5] = (__bf16)b.y; o.h[6] = (__bf16)b.z; o.h[7] = (__bf16)b.w;
  *(uint4*)(dst + i) = o.u;
}

// ---------------------------------------------------------------------------
// QKV GEMM: [4096,2048]bf16 @ [2048,6144]bf16 -> q/k/v [B,H,T,D] bf16.
// 128x128 block tile, k-step 32, 8 waves x (4x2) WMMA tiles. A tile via TDM.
// Q part pre-scaled by 1/sqrt(D).
// ---------------------------------------------------------------------------
__global__ void __launch_bounds__(256) k_gemm_qkv(const __bf16* __restrict__ X,
                                                  const __bf16* __restrict__ W,
                                                  __bf16* __restrict__ Qo,
                                                  __bf16* __restrict__ Ko,
                                                  __bf16* __restrict__ Vo) {
  constexpr int Kd = 2048, N = 6144, T = 2048, C = 2048, Dh = 128;
  __shared__ __bf16 ldsA[128 * 32];  // [m][k]
  __shared__ __bf16 ldsB[128 * 32];  // [n][k] (transposed for contiguous frags)
  const int tid = threadIdx.x;
  const int w = tid >> 5, lane = tid & 31;
  const int waveM = w & 1, waveN = w >> 1;
  const int mBase = blockIdx.y * 128;
  const int nBase = blockIdx.x * 128;

  v8f zero;
#pragma unroll
  for (int e = 0; e < 8; ++e) zero[e] = 0.0f;
  v8f acc[4][2];
#pragma unroll
  for (int i = 0; i < 4; ++i)
#pragma unroll
    for (int j = 0; j < 2; ++j) acc[i][j] = zero;

  for (int k0 = 0; k0 < Kd; k0 += 32) {
    __syncthreads();
#if defined(HAVE_TDM)
    if (tid < 32)  // wave 0: DMA the A tile (128 rows x 32 k, row stride Kd)
      tdm_load_tile_bf16(lds_offset_of(&ldsA[0]), &X[(long)mBase * Kd + k0],
                         32u, 128u, (unsigned)Kd, (unsigned)Kd, 4096u);
#else
#pragma unroll
    for (int it = 0; it < 2; ++it) {            // A tile: 128x32
      int idx = tid + it * 256;
      int row = idx >> 2;
      int kc  = (idx & 3) * 8;
      *(uint4*)&ldsA[row * 32 + kc] =
          *(const uint4*)&X[(long)(mBase + row) * Kd + k0 + kc];
    }
#endif
#pragma unroll
    for (int it = 0; it < 2; ++it) {            // B tile: 32x128, store transposed
      int idx = tid + it * 256;
      int kk = idx >> 4;
      int nc = (idx & 15) * 8;
      uint4 d = *(const uint4*)&W[(long)(k0 + kk) * N + nBase + nc];
      const __bf16* hp = (const __bf16*)&d;
#pragma unroll
      for (int j = 0; j < 8; ++j) ldsB[(nc + j) * 32 + kk] = hp[j];
    }
    if (k0 + 32 < Kd) __builtin_prefetch(&W[(long)(k0 + 32) * N + nBase], 0, 0);
#if defined(HAVE_TDM)
    if (tid < 32) tdm_wait();
#endif
    __syncthreads();

    v16bf af[4], bfg[2];
#pragma unroll
    for (int mt = 0; mt < 4; ++mt)
      af[mt] = load_frag(&ldsA[(waveM * 64 + mt * 16) * 32], 32);
#pragma unroll
    for (int nt = 0; nt < 2; ++nt)
      bfg[nt] = load_frag(&ldsB[(waveN * 32 + nt * 16) * 32], 32);
#pragma unroll
    for (int mt = 0; mt < 4; ++mt)
#pragma unroll
      for (int nt = 0; nt < 2; ++nt)
        acc[mt][nt] = WMMA_BF16(af[mt], bfg[nt], acc[mt][nt]);
  }

  // epilogue: scatter into q/k/v [B,H,T,D]; fold softmax scale into Q
  const int part = nBase >> 11;                       // 0=q,1=k,2=v (uniform per block)
  const float scl = (part == 0) ? 0.08838834764831845f : 1.0f;
  __bf16* dstBuf = (part == 0) ? Qo : ((part == 1) ? Ko : Vo);
  const int half = lane >> 4, col0 = lane & 15;
#pragma unroll
  for (int mt = 0; mt < 4; ++mt)
#pragma unroll
    for (int nt = 0; nt < 2; ++nt)
#pragma unroll
      for (int r = 0; r < 8; ++r) {
        int row  = mBase + waveM * 64 + mt * 16 + r + half * 8;  // b*T + t
        int ncol = nBase + waveN * 32 + nt * 16 + col0;
        int c = ncol & (C - 1);
        int h = c >> 7, d = c & 127;
        int b = row >> 11, t = row & (T - 1);
        dstBuf[(((long)(b * 16 + h)) * T + t) * Dh + d] = (__bf16)(acc[mt][nt][r] * scl);
      }
}

// ---------------------------------------------------------------------------
// Flash attention: one block per (b, h, 128 query rows). 8 waves x 16 rows.
// Key blocks of 64; K tile via TDM ([key][d], contiguous), V^T cooperative;
// online softmax; P re-layout (C-layout -> A-layout) through per-wave LDS.
// ---------------------------------------------------------------------------
__global__ void __launch_bounds__(256) k_attn(const __bf16* __restrict__ Qb,
                                              const __bf16* __restrict__ Kb,
                                              const __bf16* __restrict__ Vb,
                                              __bf16* __restrict__ Y) {
  constexpr int T = 2048, Dh = 128, H = 16, KB = 64, C = 2048;
  __shared__ __bf16 ldsK[KB * Dh];       // [key][d]
  __shared__ __bf16 ldsVt[Dh * KB];      // [d][key]
  __shared__ __bf16 ldsP[8 * 16 * KB];   // per-wave [m][key]
  const int tid = threadIdx.x, w = tid >> 5, lane = tid & 31;
  const int half = lane >> 4, col = lane & 15;
  const int qb = blockIdx.x & 15;
  const int h  = (blockIdx.x >> 4) & 15;
  const int b  = blockIdx.x >> 8;
  const long headOff = ((long)(b * H + h)) * T * Dh;
  const int qRow0 = qb * 128 + w * 16;

  // Q fragments straight from global (rows contiguous in D, A-layout)
  v16bf qf[4];
#pragma unroll
  for (int ds = 0; ds < 4; ++ds)
    qf[ds] = load_frag(Qb + headOff + (long)qRow0 * Dh + ds * 32, Dh);

  v8f zero;
#pragma unroll
  for (int e = 0; e < 8; ++e) zero[e] = 0.0f;
  v8f o[8];
#pragma unroll
  for (int dt = 0; dt < 8; ++dt) o[dt] = zero;
  float m_i[8], l_i[8];
#pragma unroll
  for (int r = 0; r < 8; ++r) { m_i[r] = -1e30f; l_i[r] = 0.0f; }

  const int nKB = (qb + 1) * 2;
  for (int kb = 0; kb < nKB; ++kb) {
    const int kBase = kb * KB;
    __syncthreads();
#if defined(HAVE_TDM)
    if (tid < 32)  // wave 0: DMA K tile (64 keys x 128 d, fully contiguous)
      tdm_load_tile_bf16(lds_offset_of(&ldsK[0]), Kb + headOff + (long)kBase * Dh,
                         128u, (unsigned)KB, (unsigned)Dh, (unsigned)Dh, (unsigned)T);
#endif
#pragma unroll
    for (int it = 0; it < 4; ++it) {     // V tile 64x128 -> LDS transposed
      int idx = tid + it * 256;          // 0..1023
      int row = idx >> 4;                // key 0..63
      int dc  = (idx & 15) * 8;
#if !defined(HAVE_TDM)
      uint4 kd = *(const uint4*)&Kb[headOff + (long)(kBase + row) * Dh + dc];
      *(uint4*)&ldsK[row * Dh + dc] = kd;
#endif
      uint4 vd = *(const uint4*)&Vb[headOff + (long)(kBase + row) * Dh + dc];
      const __bf16* hp = (const __bf16*)&vd;
#pragma unroll
      for (int j = 0; j < 8; ++j) ldsVt[(dc + j) * KB + row] = hp[j];
    }
#if defined(HAVE_TDM)
    if (tid < 32) tdm_wait();
#endif
    __syncthreads();

    // S = Q K^T for this wave's 16 rows vs 64 keys
    v8f s[4];
#pragma unroll
    for (int nt = 0; nt < 4; ++nt) {
      s[nt] = zero;
#pragma unroll
      for (int ds = 0; ds < 4; ++ds) {
        v16bf kf = load_frag(&ldsK[(nt * 16) * Dh + ds * 32], Dh);
        s[nt] = WMMA_BF16(qf[ds], kf, s[nt]);
      }
    }
    // causal mask (only key blocks that can cross the diagonal)
    if (kBase + KB - 1 > qRow0) {
#pragma unroll
      for (int nt = 0; nt < 4; ++nt)
#pragma unroll
        for (int r = 0; r < 8; ++r)
          if (kBase + nt * 16 + col > qRow0 + r + half * 8) s[nt][r] = -1e30f;
    }

    // online softmax: row stats live per (half-wave, r)
    float alpha[8];
#pragma unroll
    for (int r = 0; r < 8; ++r) {
      float mx = s[0][r];
#pragma unroll
      for (int nt = 1; nt < 4; ++nt) mx = fmaxf(mx, s[nt][r]);
      mx = hmax16(mx);
      float mn = fmaxf(m_i[r], mx);
      alpha[r] = __expf(m_i[r] - mn);
      m_i[r] = mn;
    }
    __bf16* Pw = &ldsP[w * 16 * KB];
    float rs[8];
#pragma unroll
    for (int r = 0; r < 8; ++r) rs[r] = 0.0f;
#pragma unroll
    for (int nt = 0; nt < 4; ++nt)
#pragma unroll
      for (int r = 0; r < 8; ++r) {
        float p = __expf(s[nt][r] - m_i[r]);
        rs[r] += p;
        Pw[(r + half * 8) * KB + nt * 16 + col] = (__bf16)p;  // C-layout -> A-layout
      }
#pragma unroll
    for (int r = 0; r < 8; ++r) l_i[r] = l_i[r] * alpha[r] + hsum16(rs[r]);
#pragma unroll
    for (int dt = 0; dt < 8; ++dt)
#pragma unroll
      for (int r = 0; r < 8; ++r) o[dt][r] *= alpha[r];

    // O += P @ V
#pragma unroll
    for (int ks = 0; ks < 2; ++ks) {
      v16bf pf = load_frag(Pw + ks * 32, KB);
#pragma unroll
      for (int dt = 0; dt < 8; ++dt) {
        v16bf vf = load_frag(&ldsVt[(dt * 16) * KB + ks * 32], KB);
        o[dt] = WMMA_BF16(pf, vf, o[dt]);
      }
    }
  }

  // normalize and store y [B,T,C] bf16
#pragma unroll
  for (int r = 0; r < 8; ++r) {
    float inv = 1.0f / l_i[r];
    int trow = qRow0 + r + half * 8;
#pragma unroll
    for (int dt = 0; dt < 8; ++dt)
      Y[((long)b * T + trow) * C + h * Dh + dt * 16 + col] = (__bf16)(o[dt][r] * inv);
  }
}

// ---------------------------------------------------------------------------
// Projection GEMM: y[4096,2048]bf16 @ w_proj[2048,2048]bf16 -> out fp32
// ---------------------------------------------------------------------------
__global__ void __launch_bounds__(256) k_gemm_proj(const __bf16* __restrict__ Yb,
                                                   const __bf16* __restrict__ W,
                                                   float* __restrict__ Out) {
  constexpr int Kd = 2048, N = 2048;
  __shared__ __bf16 ldsA[128 * 32];
  __shared__ __bf16 ldsB[128 * 32];
  const int tid = threadIdx.x;
  const int w = tid >> 5, lane = tid & 31;
  const int waveM = w & 1, waveN = w >> 1;
  const int mBase = blockIdx.y * 128;
  const int nBase = blockIdx.x * 128;

  v8f zero;
#pragma unroll
  for (int e = 0; e < 8; ++e) zero[e] = 0.0f;
  v8f acc[4][2];
#pragma unroll
  for (int i = 0; i < 4; ++i)
#pragma unroll
    for (int j = 0; j < 2; ++j) acc[i][j] = zero;

  for (int k0 = 0; k0 < Kd; k0 += 32) {
    __syncthreads();
#if defined(HAVE_TDM)
    if (tid < 32)
      tdm_load_tile_bf16(lds_offset_of(&ldsA[0]), &Yb[(long)mBase * Kd + k0],
                         32u, 128u, (unsigned)Kd, (unsigned)Kd, 4096u);
#else
#pragma unroll
    for (int it = 0; it < 2; ++it) {
      int idx = tid + it * 256;
      int row = idx >> 2;
      int kc  = (idx & 3) * 8;
      *(uint4*)&ldsA[row * 32 + kc] =
          *(const uint4*)&Yb[(long)(mBase + row) * Kd + k0 + kc];
    }
#endif
#pragma unroll
    for (int it = 0; it < 2; ++it) {
      int idx = tid + it * 256;
      int kk = idx >> 4;
      int nc = (idx & 15) * 8;
      uint4 d = *(const uint4*)&W[(long)(k0 + kk) * N + nBase + nc];
      const __bf16* hp = (const __bf16*)&d;
#pragma unroll
      for (int j = 0; j < 8; ++j) ldsB[(nc + j) * 32 + kk] = hp[j];
    }
#if defined(HAVE_TDM)
    if (tid < 32) tdm_wait();
#endif
    __syncthreads();

    v16bf af[4], bfg[2];
#pragma unroll
    for (int mt = 0; mt < 4; ++mt)
      af[mt] = load_frag(&ldsA[(waveM * 64 + mt * 16) * 32], 32);
#pragma unroll
    for (int nt = 0; nt < 2; ++nt)
      bfg[nt] = load_frag(&ldsB[(waveN * 32 + nt * 16) * 32], 32);
#pragma unroll
    for (int mt = 0; mt < 4; ++mt)
#pragma unroll
      for (int nt = 0; nt < 2; ++nt)
        acc[mt][nt] = WMMA_BF16(af[mt], bfg[nt], acc[mt][nt]);
  }

  const int half = lane >> 4, col0 = lane & 15;
#pragma unroll
  for (int mt = 0; mt < 4; ++mt)
#pragma unroll
    for (int nt = 0; nt < 2; ++nt)
#pragma unroll
      for (int r = 0; r < 8; ++r) {
        int row  = mBase + waveM * 64 + mt * 16 + r + half * 8;
        int ncol = nBase + waveN * 32 + nt * 16 + col0;
        Out[(long)row * N + ncol] = acc[mt][nt][r];
      }
}

// ---------------------------------------------------------------------------
extern "C" void kernel_launch(void* const* d_in, const int* in_sizes, int n_in,
                              void* d_out, int out_size, void* d_ws, size_t ws_size,
                              hipStream_t stream) {
  (void)in_sizes; (void)n_in; (void)out_size; (void)ws_size;
  const float* x  = (const float*)d_in[0];   // [2,2048,2048]
  const float* wa = (const float*)d_in[1];   // [2048,6144]
  const float* wp = (const float*)d_in[2];   // [2048,2048]
  float* out = (float*)d_out;                // [2,2048,2048]

  constexpr long M = 4096, Cdim = 2048, N3 = 6144;
  constexpr long HEAD_ELEMS = 2L * 16 * 2048 * 128;

  char* ws = (char*)d_ws;
  size_t off = 0;
  auto carve = [&](size_t bytes) {
    void* p = ws + off;
    off += (bytes + 255) & ~(size_t)255;
    return p;
  };
  __bf16* xbf  = (__bf16*)carve(M * Cdim * 2);
  __bf16* wabf = (__bf16*)carve(Cdim * N3 * 2);
  __bf16* wpbf = (__bf16*)carve(Cdim * Cdim * 2);
  __bf16* qbf  = (__bf16*)carve(HEAD_ELEMS * 2);
  __bf16* kbf  = (__bf16*)carve(HEAD_ELEMS * 2);
  __bf16* vbf  = (__bf16*)carve(HEAD_ELEMS * 2);
  __bf16* ybf  = (__bf16*)carve(M * Cdim * 2);

  // 1) fp32 -> bf16
  k_f32_to_bf16<<<(unsigned)((M * Cdim / 8 + 255) / 256), 256, 0, stream>>>(x, xbf, M * Cdim);
  k_f32_to_bf16<<<(unsigned)((Cdim * N3 / 8 + 255) / 256), 256, 0, stream>>>(wa, wabf, Cdim * N3);
  k_f32_to_bf16<<<(unsigned)((Cdim * Cdim / 8 + 255) / 256), 256, 0, stream>>>(wp, wpbf, Cdim * Cdim);

  // 2) QKV GEMM -> q/k/v [B,H,T,D] (q pre-scaled)
  k_gemm_qkv<<<dim3(48, 32), 256, 0, stream>>>(xbf, wabf, qbf, kbf, vbf);

  // 3) flash attention -> y [B,T,C] bf16
  k_attn<<<dim3(512), 256, 0, stream>>>(qbf, kbf, vbf, ybf);

  // 4) projection -> fp32 out
  k_gemm_proj<<<dim3(16, 32), 256, 0, stream>>>(ybf, wpbf, out);
}
